// MyPytorchGRU_87084756893719
// MI455X (gfx1250) — matrix operand order repaired and checked by reading, not compiled, not run on hardware
//
#include <hip/hip_runtime.h>

// GRU: B=32, L=2048, D=H=256, G=3H=768.
// Persistent-scan design: 16 workgroups x 192 threads (6 wave32).
// WG g owns output columns [16g,16g+16) of H, i.e. 48 gate rows
// (update/reset/new) of both W_ih and W_hh.
// Weights are loop-invariant -> each wave keeps its B-matrix WMMA fragments
// for both GEMMs register-resident (128 VGPRs) for the whole scan.
// Per step: stage x_t and h (L2-resident double buffer) to LDS as bf16,
// run v_wmma_f32_16x16x32_bf16 for both projections, do gate math in f32,
// write h_{t+1} chunk + output chunk, then device-scope barrier.

#define GRU_B 32
#define GRU_L 2048
#define GRU_D 256
#define GRU_H 256
#define GRU_NWG 16
#define GRU_NTHR 192

typedef __attribute__((ext_vector_type(16))) __bf16 v16bf;
typedef __attribute__((ext_vector_type(8)))  __bf16 v8bf;
typedef __attribute__((ext_vector_type(8)))  float  v8f;

__device__ __forceinline__ __bf16 f32_to_bf16(float f) {
    union { float f; unsigned u; } v; v.f = f;
    // round-to-nearest-even
    unsigned r = v.u + 0x7FFFu + ((v.u >> 16) & 1u);
    union { unsigned short s; __bf16 b; } o;
    o.s = (unsigned short)(r >> 16);
    return o.b;
}

__global__ void gru_init_ws(float* ws, int n) {
    int i = blockIdx.x * blockDim.x + threadIdx.x;
    if (i < n) ws[i] = 0.0f;
}

__global__ __launch_bounds__(GRU_NTHR, 1) void gru_scan(
    const float* __restrict__ x,      // [B, L, D]
    const float* __restrict__ Wih,    // [3H, D]
    const float* __restrict__ bih,    // [3H]
    const float* __restrict__ Whh,    // [3H, H]
    float* __restrict__ out,          // [B, L, H]
    float* __restrict__ hbuf,         // [2][B*H] double buffer (ws)
    unsigned* __restrict__ ctr)       // monotonic arrive counter (ws)
{
    __shared__ __align__(16) __bf16 sX[GRU_B][GRU_D];  // 16 KB
    __shared__ __align__(16) __bf16 sH[GRU_B][GRU_H];  // 16 KB
    __shared__ float sGX[GRU_B][48];                   // 6 KB
    __shared__ float sGH[GRU_B][48];                   // 6 KB

    const int tid = threadIdx.x;
    const int wg  = blockIdx.x;          // 0..15, owns H cols [16*wg, 16*wg+16)

    // Wave/tile mapping: 6 waves -> (mtile 0..1) x (ntile 0..2)
    const int wv    = tid >> 5;
    const int lane  = tid & 31;
    const int half  = lane >> 4;         // which 16-lane half
    const int l15   = lane & 15;
    const int mtile = wv & 1;
    const int ntile = wv >> 1;
    const int arow  = mtile * 16 + l15;  // batch row this lane supplies for A
    const int bcol  = ntile * 16 + l15;  // local gate col this lane supplies for B

    // ---- Register-resident B fragments for both weight matrices ----
    // Local gate col bcol -> global gate row; B[k][n] = W[grow][k], so this
    // lane needs contiguous K-slices of its own W row: pure b128 loads.
    const int gate = bcol >> 4;          // 0=update,1=reset,2=new
    const int grow = gate * GRU_H + wg * 16 + (bcol & 15);
    const float bias = bih[grow];

    v16bf wBx[8], wBh[8];                // 8 k-chunks x 2 matrices = 128 VGPRs
    #pragma unroll
    for (int c = 0; c < 8; ++c) {
        const float* px = &Wih[grow * GRU_D + c * 32 + 16 * half];
        const float* ph = &Whh[grow * GRU_H + c * 32 + 16 * half];
        v16bf fx, fh;
        #pragma unroll
        for (int q = 0; q < 4; ++q) {
            const float4 xv = *(const float4*)(px + 4 * q);
            const float4 hv = *(const float4*)(ph + 4 * q);
            fx[4 * q + 0] = f32_to_bf16(xv.x); fx[4 * q + 1] = f32_to_bf16(xv.y);
            fx[4 * q + 2] = f32_to_bf16(xv.z); fx[4 * q + 3] = f32_to_bf16(xv.w);
            fh[4 * q + 0] = f32_to_bf16(hv.x); fh[4 * q + 1] = f32_to_bf16(hv.y);
            fh[4 * q + 2] = f32_to_bf16(hv.z); fh[4 * q + 3] = f32_to_bf16(hv.w);
        }
        wBx[c] = fx;
        wBh[c] = fh;
    }

    for (int t = 0; t < GRU_L; ++t) {
        const float* hcur = hbuf + (t & 1) * (GRU_B * GRU_H);
        float*       hnxt = hbuf + ((t + 1) & 1) * (GRU_B * GRU_H);

        // Prefetch next timestep's x rows while we work on this one.
        if (tid < GRU_B && (t + 1) < GRU_L)
            __builtin_prefetch(&x[((size_t)tid * GRU_L + (t + 1)) * GRU_D], 0, 1);

        // ---- Stage x_t and h_t into LDS as bf16 (128-bit global loads) ----
        for (int idx = tid; idx < GRU_B * GRU_D / 4; idx += GRU_NTHR) {
            int b  = idx >> 6;
            int d4 = (idx & 63) << 2;
            const float4 xv = *(const float4*)&x[((b * GRU_L) + t) * GRU_D + d4];
            sX[b][d4 + 0] = f32_to_bf16(xv.x);
            sX[b][d4 + 1] = f32_to_bf16(xv.y);
            sX[b][d4 + 2] = f32_to_bf16(xv.z);
            sX[b][d4 + 3] = f32_to_bf16(xv.w);
            const float4 hv = *(const float4*)&hcur[(b << 8) + d4];
            sH[b][d4 + 0] = f32_to_bf16(hv.x);
            sH[b][d4 + 1] = f32_to_bf16(hv.y);
            sH[b][d4 + 2] = f32_to_bf16(hv.z);
            sH[b][d4 + 3] = f32_to_bf16(hv.w);
        }
        __syncthreads();

        // ---- WMMA: xgate = x_t @ Wih_slice^T + b ; hgate = h @ Whh_slice^T ----
        v8f accx, acch;
        #pragma unroll
        for (int v = 0; v < 8; ++v) { accx[v] = bias; acch[v] = 0.0f; }

        #pragma unroll
        for (int c = 0; c < 8; ++c) {
            const int kk = c * 32;
            v16bf ax, ah;
            // A fragment (16x32 bf16): lanes 0-15 hold K {0..15},
            // lanes 16-31 hold K {16..31} of this k-chunk.
            const v8bf* p0 = (const v8bf*)&sX[arow][kk + 16 * half];
            const v8bf* p1 = (const v8bf*)&sX[arow][kk + 16 * half + 8];
            const v8bf* q0 = (const v8bf*)&sH[arow][kk + 16 * half];
            const v8bf* q1 = (const v8bf*)&sH[arow][kk + 16 * half + 8];
            #pragma unroll
            for (int j = 0; j < 8; ++j) {
                ax[j] = (*p0)[j]; ax[j + 8] = (*p1)[j];
                ah[j] = (*q0)[j]; ah[j + 8] = (*q1)[j];
            }
            accx = __builtin_amdgcn_wmma_f32_16x16x32_bf16(
                false, ax, false, wBx[c], (short)0, accx, false, false);
            acch = __builtin_amdgcn_wmma_f32_16x16x32_bf16(
                false, ah, false, wBh[c], (short)0, acch, false, false);
        }
        // D tile layout: lane (l15=N), vgpr v -> M = v + 8*half
        #pragma unroll
        for (int v = 0; v < 8; ++v) {
            int m = mtile * 16 + v + 8 * half;
            sGX[m][bcol] = accx[v];
            sGH[m][bcol] = acch[v];
        }
        __syncthreads();

        // ---- Gate math + state/output update for owned 16 columns ----
        for (int idx = tid; idx < GRU_B * 16; idx += GRU_NTHR) {
            int b  = idx >> 4;
            int ic = idx & 15;
            float xu = sGX[b][ic],      hu = sGH[b][ic];
            float xr = sGX[b][16 + ic], hr = sGH[b][16 + ic];
            float xn = sGX[b][32 + ic], hn = sGH[b][32 + ic];
            float rg = 1.0f / (1.0f + __expf(-(xr + hr)));
            float zg = 1.0f / (1.0f + __expf(-(xu + hu)));
            float ng = tanhf(xn + rg * hn);
            int col = wg * 16 + ic;
            float hold = hcur[(b << 8) + col];
            float hnew = zg * hold + (1.0f - zg) * ng;
            hnxt[(b << 8) + col] = hnew;
            out[((b * GRU_L) + t) * GRU_H + col] = hnew;
        }

        // ---- Device-scope split barrier across the 16 persistent WGs ----
        __threadfence();
        __syncthreads();
        if (tid == 0) {
            __hip_atomic_fetch_add(ctr, 1u, __ATOMIC_RELEASE,
                                   __HIP_MEMORY_SCOPE_AGENT);
            unsigned target = (unsigned)GRU_NWG * (unsigned)(t + 1);
            while (__hip_atomic_load(ctr, __ATOMIC_ACQUIRE,
                                     __HIP_MEMORY_SCOPE_AGENT) < target) {
                __builtin_amdgcn_s_sleep(2);
            }
        }
        __syncthreads();
    }
}

extern "C" void kernel_launch(void* const* d_in, const int* in_sizes, int n_in,
                              void* d_out, int out_size, void* d_ws, size_t ws_size,
                              hipStream_t stream) {
    (void)in_sizes; (void)n_in; (void)out_size; (void)ws_size;
    const float* x   = (const float*)d_in[0];   // [32,2048,256]
    const float* Wih = (const float*)d_in[1];   // [768,256]
    const float* bih = (const float*)d_in[2];   // [768]
    const float* Whh = (const float*)d_in[3];   // [768,256]
    float*       out = (float*)d_out;           // [32,2048,256]

    float*    hbuf = (float*)d_ws;                                  // 2*32*256 f32
    unsigned* ctr  = (unsigned*)((char*)d_ws +
                                 2ull * GRU_B * GRU_H * sizeof(float));

    // Re-zero h double-buffer + barrier counter every launch (graph-replay safe).
    int zn = 2 * GRU_B * GRU_H + 64;
    gru_init_ws<<<(zn + 255) / 256, 256, 0, stream>>>((float*)d_ws, zn);

    gru_scan<<<GRU_NWG, GRU_NTHR, 0, stream>>>(x, Wih, bih, Whh, out, hbuf, ctr);
}
